// LEDBasicSentenceClassificationModel_24678882083277
// MI455X (gfx1250) — matrix operand order repaired and checked by reading, not compiled, not run on hardware
//
#include <hip/hip_runtime.h>
#include <math.h>

// ---------------- model constants ----------------
namespace {
constexpr int cB = 2, cS = 2048, cD = 768, cH = 12, cF = 3072;
constexpr int cNL = 3, cW = 256, cG = 64, cNS = 64, cDH = 64;
constexpr int BOS_ID = 0, PAD_ID = 1;
}

typedef __bf16 bf16;
typedef __attribute__((ext_vector_type(16))) __bf16 v16bf;
typedef __attribute__((ext_vector_type(8)))  float  v8f;
typedef int v4i __attribute__((vector_size(16)));   // payload type of the async b128 builtin

#if defined(__HIP_DEVICE_COMPILE__) && __has_builtin(__builtin_amdgcn_global_load_async_to_lds_b128)
#define HAVE_ASYNC 1
#else
#define HAVE_ASYNC 0
#endif

#if HAVE_ASYNC
#if __has_builtin(__builtin_amdgcn_s_wait_asynccnt)
#define ASYNC_WAIT() __builtin_amdgcn_s_wait_asynccnt(0)
#else
#define ASYNC_WAIT() asm volatile("s_wait_asynccnt 0x0" ::: "memory")
#endif
#else
#define ASYNC_WAIT() ((void)0)
#endif

// ---- f32 -> bf16 (RNE) bit helpers ----
static __device__ inline unsigned bfbits(float f) {
    union { float f; unsigned u; } cv; cv.f = f;
    unsigned u = cv.u;
    return ((u + 0x7FFFu + ((u >> 16) & 1u)) >> 16) & 0xFFFFu;
}
static __device__ inline unsigned pack2(float lo, float hi) {
    return bfbits(lo) | (bfbits(hi) << 16);
}

static __device__ inline float gelu_exact(float x) {
    return 0.5f * x * (1.0f + erff(x * 0.70710678118654752f));
}

// ================= conversion kernels =================
// elementwise f32 -> packed bf16, 8 elements per thread
__global__ __launch_bounds__(256)
void k_cvt(const float* __restrict__ in, ushort* __restrict__ out, int n8) {
    int i = blockIdx.x * 256 + threadIdx.x;
    if (i < n8) {
        float4 a = ((const float4*)in)[2 * i];
        float4 b = ((const float4*)in)[2 * i + 1];
        uint4 o;
        o.x = pack2(a.x, a.y);
        o.y = pack2(a.z, a.w);
        o.z = pack2(b.x, b.y);
        o.w = pack2(b.z, b.w);
        ((uint4*)out)[i] = o;
    }
}

// W[K,N] f32  ->  Wt[N,K] bf16 (transpose + convert), 32x32 LDS tiles
__global__ __launch_bounds__(256)
void k_twcvt(const float* __restrict__ in, ushort* __restrict__ out, int K, int N) {
    __shared__ float tile[32][33];
    const int n0 = blockIdx.x * 32, k0 = blockIdx.y * 32;
    const int c = threadIdx.x & 31, r8 = threadIdx.x >> 5;
#pragma unroll
    for (int i = 0; i < 4; ++i) {
        int r = r8 + i * 8;
        tile[r][c] = in[(size_t)(k0 + r) * N + n0 + c];
    }
    __syncthreads();
#pragma unroll
    for (int i = 0; i < 4; ++i) {
        int r = r8 + i * 8;   // n-offset
        out[(size_t)(n0 + r) * K + k0 + c] = (ushort)bfbits(tile[c][r]);
    }
}

// ================= WMMA GEMM =================
// C[M,N] = act(A@W + bias).  A: bf16 [M,K] row-major.  Wt: bf16 [N,K] row-major.
// Requires M%128==0, N%128==0, K%32==0 (true for all call sites).
#define TM 128
#define TN 128
#define TK 32
#define LSA 40   // bf16 elems per LDS row (80B: 16B-aligned, conflict-free)

// stage one [128 rows x 32 bf16] tile (rows contiguous in K) into LDS
static __device__ inline void stage_tile(const ushort* __restrict__ gbase, int ld,
                                         int row0, int k0, bf16* lds, int t) {
#pragma unroll
    for (int i = 0; i < 2; ++i) {
        int chunk = i * 256 + t;            // 512 x 16B chunks
        int r  = chunk >> 2;
        int c8 = (chunk & 3) << 3;          // bf16 element offset within row
        const ushort* g = gbase + (size_t)(row0 + r) * ld + k0 + c8;
        bf16* l = lds + r * LSA + c8;
#if HAVE_ASYNC
        __builtin_amdgcn_global_load_async_to_lds_b128((v4i*)g, (v4i*)l, 0, 0);
#else
        *(uint4*)l = *(const uint4*)g;
#endif
    }
}

__global__ __launch_bounds__(256)
void k_gemm(const ushort* __restrict__ A, const ushort* __restrict__ Wt,
            const float* __restrict__ bias, float* __restrict__ C,
            int M, int N, int K, int act) {
    __shared__ alignas(16) bf16 AsBuf[2][TM * LSA];
    __shared__ alignas(16) bf16 BsBuf[2][TN * LSA];

    const int t    = threadIdx.x;
    const int wave = t >> 5;
    const int lane = t & 31;
    const int bm = blockIdx.y * TM;
    const int bn = blockIdx.x * TN;
    const int wm = (wave >> 2) * 64;     // 0 / 64
    const int wn = (wave & 3) * 32;      // 0 / 32 / 64 / 96
    const int lr = lane & 15;
    const int lh = lane >> 4;

    v8f acc[4][2];
#pragma unroll
    for (int m = 0; m < 4; ++m)
#pragma unroll
        for (int n = 0; n < 2; ++n) {
            v8f z = {0.f, 0.f, 0.f, 0.f, 0.f, 0.f, 0.f, 0.f};
            acc[m][n] = z;
        }

    // prologue: stage first K-tile
    stage_tile(A,  K, bm, 0, &AsBuf[0][0], t);
    stage_tile(Wt, K, bn, 0, &BsBuf[0][0], t);
    ASYNC_WAIT();
    __syncthreads();

    int buf = 0;
    for (int k0 = 0; k0 < K; k0 += TK) {
        const int nxt = buf ^ 1;
        if (k0 + TK < K) {                 // overlap: fetch next tile while computing
            stage_tile(A,  K, bm, k0 + TK, &AsBuf[nxt][0], t);
            stage_tile(Wt, K, bn, k0 + TK, &BsBuf[nxt][0], t);
        }

        const bf16* As = &AsBuf[buf][0];
        const bf16* Bs = &BsBuf[buf][0];

        // ---- fragments (ISA 7.12.2 layouts) ----
        v16bf afr[4], bfr[2];
#pragma unroll
        for (int m = 0; m < 4; ++m) {
            const bf16* base = As + (wm + m * 16 + lr) * LSA;
            const int kb = lh * 8;
#pragma unroll
            for (int v = 0; v < 8; ++v) {
                int kk = ((v >> 2) << 4) + kb + ((v & 3) << 1);
                afr[m][2 * v]     = base[kk];
                afr[m][2 * v + 1] = base[kk + 1];
            }
        }
#pragma unroll
        for (int n = 0; n < 2; ++n) {
            const bf16* base = Bs + (wn + n * 16 + lr) * LSA;
#pragma unroll
            for (int v = 0; v < 8; ++v) {
                int kk = ((v >> 2) << 4) + lh * 8 + ((v & 3) << 1);
                bfr[n][2 * v]     = base[kk];
                bfr[n][2 * v + 1] = base[kk + 1];
            }
        }
        // ---- 8 WMMAs per K-step (EXEC all-ones) ----
#pragma unroll
        for (int m = 0; m < 4; ++m)
#pragma unroll
            for (int n = 0; n < 2; ++n)
                acc[m][n] = __builtin_amdgcn_wmma_f32_16x16x32_bf16(
                    false, afr[m], false, bfr[n], (short)0, acc[m][n], false, false);

        ASYNC_WAIT();
        __syncthreads();
        buf = nxt;
    }

    // ---- epilogue: C row = lh*8 + i, col = lr within each 16x16 tile ----
#pragma unroll
    for (int m = 0; m < 4; ++m)
#pragma unroll
        for (int n = 0; n < 2; ++n)
#pragma unroll
            for (int i = 0; i < 8; ++i) {
                int r = bm + wm + m * 16 + lh * 8 + i;
                int c = bn + wn + n * 16 + lr;
                float v = acc[m][n][i];
                if (bias) v += bias[c];
                if (act == 1) v = gelu_exact(v);
                C[(size_t)r * N + c] = v;
            }
}

// ================= LayerNorm kernels =================
__global__ __launch_bounds__(256)
void k_embed_ln(const int* __restrict__ src, const float* __restrict__ emb,
                const float* __restrict__ pos, const float* __restrict__ g,
                const float* __restrict__ bt, float* __restrict__ out, int dec) {
    const int s = blockIdx.x, b = blockIdx.y, t = threadIdx.x;
    __shared__ float red[256];
    int tok;
    if (dec) {
        tok = (s == 0) ? BOS_ID : src[b * cS + s - 1];
        if (tok == -100) tok = PAD_ID;
    } else {
        tok = src[b * cS + s];
    }
    size_t base = ((size_t)b * cS + s) * cD;
    float v[3]; float sum = 0.f;
#pragma unroll
    for (int i = 0; i < 3; ++i) {
        int c = t + i * 256;
        v[i] = emb[(size_t)tok * cD + c] + pos[(size_t)s * cD + c];
        sum += v[i];
    }
    red[t] = sum; __syncthreads();
    for (int off = 128; off > 0; off >>= 1) { if (t < off) red[t] += red[t + off]; __syncthreads(); }
    float mean = red[0] / (float)cD; __syncthreads();
    float sq = 0.f;
#pragma unroll
    for (int i = 0; i < 3; ++i) { float d = v[i] - mean; sq += d * d; }
    red[t] = sq; __syncthreads();
    for (int off = 128; off > 0; off >>= 1) { if (t < off) red[t] += red[t + off]; __syncthreads(); }
    float inv = rsqrtf(red[0] / (float)cD + 1e-5f);
#pragma unroll
    for (int i = 0; i < 3; ++i) {
        int c = t + i * 256;
        out[base + c] = (v[i] - mean) * inv * g[c] + bt[c];
    }
}

__global__ __launch_bounds__(256)
void k_add_ln(const float* __restrict__ x, const float* __restrict__ a,
              const float* __restrict__ g, const float* __restrict__ bt,
              float* __restrict__ out) {
    const int s = blockIdx.x, b = blockIdx.y, t = threadIdx.x;
    __shared__ float red[256];
    size_t base = ((size_t)b * gridDim.x + s) * cD;
    float v[3]; float sum = 0.f;
#pragma unroll
    for (int i = 0; i < 3; ++i) {
        int c = t + i * 256;
        v[i] = x[base + c] + a[base + c];
        sum += v[i];
    }
    red[t] = sum; __syncthreads();
    for (int off = 128; off > 0; off >>= 1) { if (t < off) red[t] += red[t + off]; __syncthreads(); }
    float mean = red[0] / (float)cD; __syncthreads();
    float sq = 0.f;
#pragma unroll
    for (int i = 0; i < 3; ++i) { float d = v[i] - mean; sq += d * d; }
    red[t] = sq; __syncthreads();
    for (int off = 128; off > 0; off >>= 1) { if (t < off) red[t] += red[t + off]; __syncthreads(); }
    float inv = rsqrtf(red[0] / (float)cD + 1e-5f);
#pragma unroll
    for (int i = 0; i < 3; ++i) {
        int c = t + i * 256;
        out[base + c] = (v[i] - mean) * inv * g[c] + bt[c];
    }
}

// ================= attention (wave32 online-softmax) =================
static __device__ inline void attn_step(float q0, float q1,
                                        const float* __restrict__ Kr,
                                        const float* __restrict__ Vr,
                                        float ok, int lane,
                                        float& m, float& l, float& a0, float& a1) {
    float p = q0 * Kr[2 * lane] + q1 * Kr[2 * lane + 1];
#pragma unroll
    for (int off = 16; off > 0; off >>= 1) p += __shfl_xor(p, off, 32);
    float s = (ok > 0.5f) ? p : -1.0e9f;
    float mn = fmaxf(m, s);
    float ex = __expf(m - mn);
    float w  = __expf(s - mn);
    a0 = a0 * ex + w * Vr[2 * lane];
    a1 = a1 * ex + w * Vr[2 * lane + 1];
    l  = l * ex + w;
    m  = mn;
}

__global__ __launch_bounds__(32)
void k_attn(const float* __restrict__ Q, const float* __restrict__ K,
            const float* __restrict__ V, float* __restrict__ O,
            int Sq, int Sk, int causal, float scale) {
    const int qi = blockIdx.x, h = blockIdx.y, b = blockIdx.z, lane = threadIdx.x;
    size_t qo = ((size_t)b * Sq + qi) * cD + h * cDH;
    float q0 = Q[qo + 2 * lane] * scale, q1 = Q[qo + 2 * lane + 1] * scale;
    float m = -1e30f, l = 0.f, a0 = 0.f, a1 = 0.f;
    int kend = causal ? (qi + 1) : Sk;
    for (int k = 0; k < kend; ++k) {
        size_t ko = ((size_t)b * Sk + k) * cD + h * cDH;
        attn_step(q0, q1, K + ko, V + ko, 1.0f, lane, m, l, a0, a1);
    }
    O[qo + 2 * lane]     = a0 / l;
    O[qo + 2 * lane + 1] = a1 / l;
}

__global__ __launch_bounds__(32)
void k_attn_long(const float* __restrict__ Q, const float* __restrict__ K,
                 const float* __restrict__ V, const float* __restrict__ Qg,
                 const float* __restrict__ Kg, const float* __restrict__ Vg,
                 const float* __restrict__ key_ok, float* __restrict__ O, float scale) {
    const int qi = blockIdx.x, h = blockIdx.y, b = blockIdx.z, lane = threadIdx.x;
    size_t ro = ((size_t)b * cS + qi) * cD + h * cDH;
    float m = -1e30f, l = 0.f, a0 = 0.f, a1 = 0.f;
    if (qi < cG) {
        float q0 = Qg[ro + 2 * lane] * scale, q1 = Qg[ro + 2 * lane + 1] * scale;
        for (int k = 0; k < cS; ++k) {
            size_t ko = ((size_t)b * cS + k) * cD + h * cDH;
            attn_step(q0, q1, Kg + ko, Vg + ko, key_ok[b * cS + k], lane, m, l, a0, a1);
        }
    } else {
        float q0 = Q[ro + 2 * lane] * scale, q1 = Q[ro + 2 * lane + 1] * scale;
        for (int k = 0; k < cG; ++k) {
            size_t ko = ((size_t)b * cS + k) * cD + h * cDH;
            attn_step(q0, q1, K + ko, V + ko, key_ok[b * cS + k], lane, m, l, a0, a1);
        }
        int lo = qi - cW; if (lo < cG) lo = cG;
        int hi = qi + cW; if (hi > cS - 1) hi = cS - 1;
        for (int k = lo; k <= hi; ++k) {
            size_t ko = ((size_t)b * cS + k) * cD + h * cDH;
            attn_step(q0, q1, K + ko, V + ko, key_ok[b * cS + k], lane, m, l, a0, a1);
        }
    }
    O[ro + 2 * lane]     = a0 / l;
    O[ro + 2 * lane + 1] = a1 / l;
}

// ================= classifier =================
__global__ __launch_bounds__(256)
void k_gather_sents(const float* __restrict__ y, const int* __restrict__ clss,
                    const float* __restrict__ mask_cls, float* __restrict__ sents) {
    const int i = blockIdx.x, b = blockIdx.y, t = threadIdx.x;
    int pos = clss[b * cNS + i];
    float mk = mask_cls[b * cNS + i];
    size_t srcb = ((size_t)b * cS + pos) * cD;
    size_t dstb = ((size_t)b * cNS + i) * cD;
#pragma unroll
    for (int j = 0; j < 3; ++j) {
        int c = t + j * 256;
        sents[dstb + c] = y[srcb + c] * mk;
    }
}

__global__ __launch_bounds__(256)
void k_cls_out(const float* __restrict__ s1, const float* __restrict__ w2,
               const float* __restrict__ b2, const float* __restrict__ mask_cls,
               float* __restrict__ out) {
    const int idx = blockIdx.x, t = threadIdx.x;
    __shared__ float red[256];
    float sum = 0.f;
    size_t base = (size_t)idx * cF;
#pragma unroll
    for (int j = 0; j < 12; ++j) {
        int c = t + j * 256;
        sum += s1[base + c] * w2[c];
    }
    red[t] = sum; __syncthreads();
    for (int off = 128; off > 0; off >>= 1) { if (t < off) red[t] += red[t + off]; __syncthreads(); }
    if (t == 0) {
        out[idx]            = (red[0] + b2[0]) * mask_cls[idx];
        out[cB * cNS + idx] = mask_cls[idx];
    }
}

// ================= host orchestration =================
static inline void cvtA(hipStream_t st, const float* A, ushort* aBf, size_t n) {
    int n8 = (int)(n / 8);
    k_cvt<<<dim3((n8 + 255) / 256), 256, 0, st>>>(A, aBf, n8);
}
static inline void gemm_pre(hipStream_t st, const ushort* aBf, const float* W, ushort* wBf,
                            const float* bias, float* C, int M, int N, int K, int act) {
    k_twcvt<<<dim3(N / 32, K / 32), 256, 0, st>>>(W, wBf, K, N);
    dim3 grid(N / TN, M / TM);
    k_gemm<<<grid, 256, 0, st>>>(aBf, wBf, bias, C, M, N, K, act);
}

extern "C" void kernel_launch(void* const* d_in, const int* in_sizes, int n_in,
                              void* d_out, int out_size, void* d_ws, size_t ws_size,
                              hipStream_t stream) {
    (void)in_sizes; (void)n_in; (void)out_size; (void)ws_size;
    const int*   src        = (const int*)  d_in[0];
    const int*   clss       = (const int*)  d_in[3];
    const float* mask_src   = (const float*)d_in[4];
    const float* mask_cls   = (const float*)d_in[5];
    const float* emb        = (const float*)d_in[6];
    const float* pos        = (const float*)d_in[7];
    const float* e_lne_g    = (const float*)d_in[8];
    const float* e_lne_b    = (const float*)d_in[9];
    const float* e_attn_w   = (const float*)d_in[10];
    const float* e_attn_b   = (const float*)d_in[11];
    const float* e_glob_w   = (const float*)d_in[12];
    const float* e_glob_b   = (const float*)d_in[13];
    const float* e_ln1_g    = (const float*)d_in[14];
    const float* e_ln1_b    = (const float*)d_in[15];
    const float* e_ln2_g    = (const float*)d_in[16];
    const float* e_ln2_b    = (const float*)d_in[17];
    const float* e_ffn_w1   = (const float*)d_in[18];
    const float* e_ffn_b1   = (const float*)d_in[19];
    const float* e_ffn_w2   = (const float*)d_in[20];
    const float* e_ffn_b2   = (const float*)d_in[21];
    const float* d_lne_g    = (const float*)d_in[22];
    const float* d_lne_b    = (const float*)d_in[23];
    const float* d_self_w   = (const float*)d_in[24];
    const float* d_self_b   = (const float*)d_in[25];
    const float* d_cross_w  = (const float*)d_in[26];
    const float* d_cross_b  = (const float*)d_in[27];
    const float* d_ln1_g    = (const float*)d_in[28];
    const float* d_ln1_b    = (const float*)d_in[29];
    const float* d_ln2_g    = (const float*)d_in[30];
    const float* d_ln2_b    = (const float*)d_in[31];
    const float* d_ln3_g    = (const float*)d_in[32];
    const float* d_ln3_b    = (const float*)d_in[33];
    const float* d_ffn_w1   = (const float*)d_in[34];
    const float* d_ffn_b1   = (const float*)d_in[35];
    const float* d_ffn_w2   = (const float*)d_in[36];
    const float* d_ffn_b2   = (const float*)d_in[37];
    const float* cls_w1     = (const float*)d_in[38];
    const float* cls_b1     = (const float*)d_in[39];
    const float* cls_w2     = (const float*)d_in[40];
    const float* cls_b2     = (const float*)d_in[41];
    float* out = (float*)d_out;

    // ---- workspace carve ----
    float* p = (float*)d_ws;
    auto take = [&](size_t n) { float* r = p; p += n; return r; };
    const size_t BSD = (size_t)cB * cS * cD;
    const size_t BSF = (size_t)cB * cS * cF;
    float* x    = take(BSD);   // encoder stream / enc_out
    float* y    = take(BSD);   // decoder stream
    float* bQ   = take(BSD);
    float* bK   = take(BSD);
    float* bV   = take(BSD);
    float* bQg  = take(BSD);
    float* bKg  = take(BSD);
    float* bVg  = take(BSD);
    float* bO   = take(BSD);
    float* bA   = take(BSD);
    float* bHf  = take(BSF);
    float* sents = take((size_t)cB * cNS * cD);
    float* s1    = take((size_t)cB * cNS * cF);
    ushort* aBf = (ushort*)p; p += BSF / 2;                    // bf16 activations (max M*K)
    ushort* wBf = (ushort*)p; p += (size_t)cD * cF / 2;        // bf16 transposed weights

    const int BS = cB * cS;
    const float sc = 0.125f;
    const size_t DD = (size_t)cD * cD, DF = (size_t)cD * cF, FD = (size_t)cF * cD;

    dim3 gRow(cS, cB), gAttn(cS, cH, cB);

    // ===== encoder =====
    k_embed_ln<<<gRow, 256, 0, stream>>>(src, emb, pos, e_lne_g, e_lne_b, x, 0);
    for (int l = 0; l < cNL; ++l) {
        const float* w  = e_attn_w + (size_t)l * 4 * DD;
        const float* bb = e_attn_b + (size_t)l * 4 * cD;
        const float* gw = e_glob_w + (size_t)l * 3 * DD;
        const float* gb = e_glob_b + (size_t)l * 3 * cD;
        cvtA(stream, x, aBf, BSD);
        gemm_pre(stream, aBf, w + 0 * DD, wBf, bb + 0 * cD, bQ,  BS, cD, cD, 0);
        gemm_pre(stream, aBf, w + 1 * DD, wBf, bb + 1 * cD, bK,  BS, cD, cD, 0);
        gemm_pre(stream, aBf, w + 2 * DD, wBf, bb + 2 * cD, bV,  BS, cD, cD, 0);
        gemm_pre(stream, aBf, gw + 0 * DD, wBf, gb + 0 * cD, bQg, BS, cD, cD, 0);
        gemm_pre(stream, aBf, gw + 1 * DD, wBf, gb + 1 * cD, bKg, BS, cD, cD, 0);
        gemm_pre(stream, aBf, gw + 2 * DD, wBf, gb + 2 * cD, bVg, BS, cD, cD, 0);
        k_attn_long<<<gAttn, 32, 0, stream>>>(bQ, bK, bV, bQg, bKg, bVg, mask_src, bO, sc);
        cvtA(stream, bO, aBf, BSD);
        gemm_pre(stream, aBf, w + 3 * DD, wBf, bb + 3 * cD, bA, BS, cD, cD, 0);
        k_add_ln<<<gRow, 256, 0, stream>>>(x, bA, e_ln1_g + l * cD, e_ln1_b + l * cD, x);
        cvtA(stream, x, aBf, BSD);
        gemm_pre(stream, aBf, e_ffn_w1 + (size_t)l * DF, wBf, e_ffn_b1 + (size_t)l * cF, bHf, BS, cF, cD, 1);
        cvtA(stream, bHf, aBf, BSF);
        gemm_pre(stream, aBf, e_ffn_w2 + (size_t)l * FD, wBf, e_ffn_b2 + (size_t)l * cD, bA, BS, cD, cF, 0);
        k_add_ln<<<gRow, 256, 0, stream>>>(x, bA, e_ln2_g + l * cD, e_ln2_b + l * cD, x);
    }
    // x is enc_out from here on

    // ===== decoder =====
    k_embed_ln<<<gRow, 256, 0, stream>>>(src, emb, pos, d_lne_g, d_lne_b, y, 1);
    for (int l = 0; l < cNL; ++l) {
        const float* w  = d_self_w + (size_t)l * 4 * DD;
        const float* bb = d_self_b + (size_t)l * 4 * cD;
        cvtA(stream, y, aBf, BSD);
        gemm_pre(stream, aBf, w + 0 * DD, wBf, bb + 0 * cD, bQ, BS, cD, cD, 0);
        gemm_pre(stream, aBf, w + 1 * DD, wBf, bb + 1 * cD, bK, BS, cD, cD, 0);
        gemm_pre(stream, aBf, w + 2 * DD, wBf, bb + 2 * cD, bV, BS, cD, cD, 0);
        k_attn<<<gAttn, 32, 0, stream>>>(bQ, bK, bV, bO, cS, cS, 1, sc);
        cvtA(stream, bO, aBf, BSD);
        gemm_pre(stream, aBf, w + 3 * DD, wBf, bb + 3 * cD, bA, BS, cD, cD, 0);
        k_add_ln<<<gRow, 256, 0, stream>>>(y, bA, d_ln1_g + l * cD, d_ln1_b + l * cD, y);

        const float* cw  = d_cross_w + (size_t)l * 4 * DD;
        const float* cbb = d_cross_b + (size_t)l * 4 * cD;
        cvtA(stream, y, aBf, BSD);
        gemm_pre(stream, aBf, cw + 0 * DD, wBf, cbb + 0 * cD, bQ, BS, cD, cD, 0);
        cvtA(stream, x, aBf, BSD);
        gemm_pre(stream, aBf, cw + 1 * DD, wBf, cbb + 1 * cD, bK, BS, cD, cD, 0);
        gemm_pre(stream, aBf, cw + 2 * DD, wBf, cbb + 2 * cD, bV, BS, cD, cD, 0);
        k_attn<<<gAttn, 32, 0, stream>>>(bQ, bK, bV, bO, cS, cS, 0, sc);
        cvtA(stream, bO, aBf, BSD);
        gemm_pre(stream, aBf, cw + 3 * DD, wBf, cbb + 3 * cD, bA, BS, cD, cD, 0);
        k_add_ln<<<gRow, 256, 0, stream>>>(y, bA, d_ln2_g + l * cD, d_ln2_b + l * cD, y);

        cvtA(stream, y, aBf, BSD);
        gemm_pre(stream, aBf, d_ffn_w1 + (size_t)l * DF, wBf, d_ffn_b1 + (size_t)l * cF, bHf, BS, cF, cD, 1);
        cvtA(stream, bHf, aBf, BSF);
        gemm_pre(stream, aBf, d_ffn_w2 + (size_t)l * FD, wBf, d_ffn_b2 + (size_t)l * cD, bA, BS, cD, cF, 0);
        k_add_ln<<<gRow, 256, 0, stream>>>(y, bA, d_ln3_g + l * cD, d_ln3_b + l * cD, y);
    }

    // ===== classifier head =====
    k_gather_sents<<<dim3(cNS, cB), 256, 0, stream>>>(y, clss, mask_cls, sents);
    cvtA(stream, sents, aBf, (size_t)cB * cNS * cD);
    gemm_pre(stream, aBf, cls_w1, wBf, cls_b1, s1, cB * cNS, cF, cD, 0);
    k_cls_out<<<cB * cNS, 256, 0, stream>>>(s1, cls_w2, cls_b2, mask_cls, out);
}
